// UltimateGSA_31817117729257
// MI455X (gfx1250) — matrix operand order repaired and checked by reading, not compiled, not run on hardware
//
#include <hip/hip_runtime.h>

// ---------------------------------------------------------------------------
// Gaussian-Splat Attention for MI455X (gfx1250, wave32, WMMA)
// ---------------------------------------------------------------------------

#define Bn   2
#define Tn   2048
#define DIMn 1024
#define Hn   8
#define Sn   12
#define HDn  128
#define MTn  (Bn * Tn)        // 4096 token rows
#define N3n  (3 * DIMn)       // 3072 qkv channels

typedef __attribute__((ext_vector_type(16))) __bf16 v16bf;
typedef __attribute__((ext_vector_type(8)))  float  v8f;

// ----------------------------- helpers -------------------------------------

__device__ __forceinline__ unsigned short f2bf_bits(float f) {
    unsigned int u = __float_as_uint(f);
    if ((u & 0x7fffffffu) > 0x7f800000u)          // NaN -> quiet NaN
        return (unsigned short)((u >> 16) | 0x0040u);
    unsigned int r = u + 0x7fffu + ((u >> 16) & 1u);  // round-nearest-even
    return (unsigned short)(r >> 16);
}

__device__ __forceinline__ float wave_sum(float v) {
    #pragma unroll
    for (int o = 16; o > 0; o >>= 1) v += __shfl_xor(v, o, 32);
    return v;
}

// ----------------------------- prep ----------------------------------------
// centers = base + delta * sigmoid(movement)*0.2 ; inv_var, c_sq, amps, 1/temp
__global__ __launch_bounds__(256) void prep_kernel(
    const float* __restrict__ base_centers, const float* __restrict__ deltas,
    const float* __restrict__ log_scales,   const float* __restrict__ log_amps,
    const float* __restrict__ movement,     const float* __restrict__ temperature,
    float* __restrict__ centers, float* __restrict__ c_sq,
    float* __restrict__ inv_var, float* __restrict__ amps,
    float* __restrict__ scal)
{
    __shared__ float ms;
    int tid = threadIdx.x;
    if (tid == 0) {
        float m = movement[0];
        ms = (1.0f / (1.0f + __expf(-m))) * 0.2f;
        scal[0] = ms;
        scal[1] = 1.0f / fmaxf(temperature[0], 0.1f);
    }
    __syncthreads();
    const int total = Hn * Sn * HDn;
    for (int i = tid; i < total; i += 256)
        centers[i] = base_centers[i] + deltas[i] * ms;
    __syncthreads();
    for (int hs = tid; hs < Hn * Sn; hs += 256) {
        float sc = __expf(log_scales[hs]);
        sc = fminf(fmaxf(sc, 1e-4f), 3.0f);
        inv_var[hs] = 0.5f / (sc * sc + 1e-8f);
        float a = __expf(log_amps[hs]);
        amps[hs] = fminf(fmaxf(a, 1e-6f), 10.0f);
        float acc = 0.0f;
        const float* cp = centers + hs * HDn;
        for (int d = 0; d < HDn; ++d) acc += cp[d] * cp[d];
        c_sq[hs] = acc;
    }
}

// --------------------------- fp32 -> bf16 -----------------------------------
__global__ __launch_bounds__(256) void f32_to_bf16_kernel(
    const float* __restrict__ src, unsigned short* __restrict__ dst, long long n)
{
    long long i = (long long)blockIdx.x * blockDim.x + threadIdx.x;
    long long stride = (long long)gridDim.x * blockDim.x;
    for (; i < n; i += stride) dst[i] = f2bf_bits(src[i]);
}

// ---------------------- batched bf16 WMMA GEMM (NT) -------------------------
// C[z](M,N) = A[z](M,K) * B[z](N,K)^T , fp32 accumulate/output.
// Block = 128 threads = 4 waves; block tile = 128(M) x 64(N);
// each wave owns a 32(M) x 64(N) register tile: 2 A-frags x 4 B-frags
// -> 8 x v_wmma_f32_16x16x32_bf16 per K-step of 32, 12 x b128 loads.
// Fragment layouts used here (contiguous k, same order in A and B):
//   A: lane L holds row M=L%16; elems [0..15]=K(kb..kb+15)
//      with kb = (L/16)*16
//   B: lane L holds row N=L%16 of B (== column of B^T); elems = K(kb16..+15)
//      with kb16 = (L/16)*16
//   C: v8f, M = r + 8*(L/16), N = L%16
__global__ __launch_bounds__(128) void gemm_nt_bf16(
    const __bf16* __restrict__ A, int lda, long long strideA,
    const __bf16* __restrict__ Bm, int ldb, long long strideB,
    float* __restrict__ C, int ldc, int cDiv, long long sC1, long long sC2,
    int M, int N, int K)
{
    const int lane = threadIdx.x & 31;
    const int wave = threadIdx.x >> 5;
    const int tileN = blockIdx.x * 64;               // block N span
    const int tileM = blockIdx.y * 128 + wave * 32;  // wave M span
    const int z = blockIdx.z;
    if (tileM >= M || tileN >= N) return;            // wave-uniform

    const __bf16* Az = A + (long long)z * strideA;
    const __bf16* Bz = Bm + (long long)z * strideB;
    float* Cz = C + (long long)(z / cDiv) * sC1 + (long long)(z % cDiv) * sC2;

    const int l15 = lane & 15;
    const int kbA = (lane >> 4) * 16;
    const int kbB = (lane >> 4) * 16;

    const __bf16* ap0 = Az + (long long)(tileM + l15)      * lda + kbA;
    const __bf16* ap1 = Az + (long long)(tileM + 16 + l15) * lda + kbA;
    const __bf16* bp0 = Bz + (long long)(tileN + l15)      * ldb + kbB;
    const long long ldb16 = (long long)ldb * 16;

    v8f acc[2][4] = {};
    for (int k0 = 0; k0 < K; k0 += 32) {
        v16bf a0, a1, b0, b1, b2, b3;
        #pragma unroll
        for (int e = 0; e < 8; ++e) {
            a0[e] = ap0[e]; a0[e + 8] = ap0[e + 8];
            a1[e] = ap1[e]; a1[e + 8] = ap1[e + 8];
        }
        #pragma unroll
        for (int e = 0; e < 16; ++e) {
            b0[e] = bp0[e];
            b1[e] = bp0[e + ldb16];
            b2[e] = bp0[e + 2 * ldb16];
            b3[e] = bp0[e + 3 * ldb16];
        }
        if (k0 + 32 < K) {                 // prefetch next K-tile (near caches)
            __builtin_prefetch(ap0 + 32, 0, 3);
            __builtin_prefetch(ap1 + 32, 0, 3);
            __builtin_prefetch(bp0 + 32, 0, 3);
        }
        acc[0][0] = __builtin_amdgcn_wmma_f32_16x16x32_bf16(false, a0, false, b0, (short)0, acc[0][0], false, false);
        acc[0][1] = __builtin_amdgcn_wmma_f32_16x16x32_bf16(false, a0, false, b1, (short)0, acc[0][1], false, false);
        acc[0][2] = __builtin_amdgcn_wmma_f32_16x16x32_bf16(false, a0, false, b2, (short)0, acc[0][2], false, false);
        acc[0][3] = __builtin_amdgcn_wmma_f32_16x16x32_bf16(false, a0, false, b3, (short)0, acc[0][3], false, false);
        acc[1][0] = __builtin_amdgcn_wmma_f32_16x16x32_bf16(false, a1, false, b0, (short)0, acc[1][0], false, false);
        acc[1][1] = __builtin_amdgcn_wmma_f32_16x16x32_bf16(false, a1, false, b1, (short)0, acc[1][1], false, false);
        acc[1][2] = __builtin_amdgcn_wmma_f32_16x16x32_bf16(false, a1, false, b2, (short)0, acc[1][2], false, false);
        acc[1][3] = __builtin_amdgcn_wmma_f32_16x16x32_bf16(false, a1, false, b3, (short)0, acc[1][3], false, false);
        ap0 += 32; ap1 += 32; bp0 += 32;
    }
    #pragma unroll
    for (int i = 0; i < 2; ++i) {
        #pragma unroll
        for (int n = 0; n < 4; ++n) {
            #pragma unroll
            for (int r = 0; r < 8; ++r) {
                int m = tileM + i * 16 + r + (lane >> 4) * 8;
                Cz[(long long)m * ldc + tileN + n * 16 + l15] = acc[i][n][r];
            }
        }
    }
}

// -------------------------- splat weights -----------------------------------
// One wave per (b,t,h): qw[bh][t][s] = exp(-max(|q-c|^2,0)*inv_var)
//                       kwa[bh][s][t] = amps * exp(-max(|k-c|^2,0)*inv_var)
__global__ __launch_bounds__(128) void splat_kernel(
    const float* __restrict__ qkv,       // [B*T, 3*DIM]
    const float* __restrict__ centers,   // [H,S,hd]
    const float* __restrict__ c_sq,      // [H,S]
    const float* __restrict__ inv_var,   // [H,S]
    const float* __restrict__ amps,      // [H,S]
    float* __restrict__ qw,              // [B*H, T, S]
    float* __restrict__ kwa)             // [B*H, S, T]
{
    int wid  = blockIdx.x * (blockDim.x >> 5) + (threadIdx.x >> 5);
    int lane = threadIdx.x & 31;
    int h = wid % Hn;
    int t = (wid / Hn) % Tn;
    int b = wid / (Hn * Tn);
    if (b >= Bn) return;
    int bh = b * Hn + h;

    const float* qp = qkv + (long long)(b * Tn + t) * N3n + h * HDn;
    const float* kp = qp + DIMn;

    float q0 = qp[lane], q1 = qp[lane + 32], q2 = qp[lane + 64], q3 = qp[lane + 96];
    float k0 = kp[lane], k1 = kp[lane + 32], k2 = kp[lane + 64], k3 = kp[lane + 96];

    float tq = wave_sum(q0 * q0 + q1 * q1 + q2 * q2 + q3 * q3);
    float tk = wave_sum(k0 * k0 + k1 * k1 + k2 * k2 + k3 * k3);

    float myq = 0.0f, myk = 0.0f;        // lane s keeps splat s's result
    for (int s = 0; s < Sn; ++s) {
        const float* cp = centers + (h * Sn + s) * HDn;
        float c0 = cp[lane], c1 = cp[lane + 32], c2 = cp[lane + 64], c3 = cp[lane + 96];
        float xq = wave_sum(q0 * c0 + q1 * c1 + q2 * c2 + q3 * c3);
        float xk = wave_sum(k0 * c0 + k1 * c1 + k2 * c2 + k3 * c3);
        float cs = c_sq[h * Sn + s];
        float iv = inv_var[h * Sn + s];
        float d2q = fmaxf(tq + cs - 2.0f * xq, 0.0f);
        float d2k = fmaxf(tk + cs - 2.0f * xk, 0.0f);
        float wq = __expf(-d2q * iv);
        float wk = __expf(-d2k * iv) * amps[h * Sn + s];
        if (lane == s) { myq = wq; myk = wk; }
    }
    if (lane < Sn) {
        qw[((long long)bh * Tn + t) * Sn + lane] = myq;
        kwa[((long long)bh * Sn + lane) * Tn + t] = myk;
    }
}

// ------- v transpose + bf16 via LDS tiles: vt[b,h,d,j] = v[b,j,h,d] ----------
// grid: (T/32, (HD/32) * B*H), block (32, 8); coalesced on both sides.
__global__ __launch_bounds__(256) void vt_transpose_kernel(
    const float* __restrict__ qkv, unsigned short* __restrict__ vt)
{
    __shared__ unsigned short tile[32][33];
    int jt = blockIdx.x * 32;
    int rest = blockIdx.y;
    int dt = (rest % (HDn / 32)) * 32; rest /= (HDn / 32);
    int h = rest % Hn;
    int b = rest / Hn;
    int tx = threadIdx.x, ty = threadIdx.y;

    #pragma unroll
    for (int u = 0; u < 4; ++u) {
        int j = jt + ty + u * 8;
        float v = qkv[(long long)(b * Tn + j) * N3n + 2 * DIMn + h * HDn + dt + tx];
        tile[ty + u * 8][tx] = f2bf_bits(v);
    }
    __syncthreads();
    #pragma unroll
    for (int u = 0; u < 4; ++u) {
        int d = dt + ty + u * 8;
        vt[((long long)(b * Hn + h) * HDn + d) * Tn + jt + tx] = tile[tx][ty + u * 8];
    }
}

// ---------------- fused rank-S logits + softmax ------------------------------
// One 256-thread block per (b,h,i).  logit[j] = invT * sum_s qw[s]*kwa[s][j].
// Writes fp32 attn (b,i,j,h layout) into d_out and bf16 attn [bh,i,j] to ws.
__global__ __launch_bounds__(256) void attn_softmax_kernel(
    const float* __restrict__ qw,    // [B*H, T, S]
    const float* __restrict__ kwa,   // [B*H, S, T]
    const float* __restrict__ scal,  // scal[1] = 1/max(temp,0.1)
    float* __restrict__ attn_out,    // [B, T, T, H]
    unsigned short* __restrict__ attn_bf)  // [B*H, T, T]
{
    const int JPT = Tn / 256;        // 8 columns per thread
    __shared__ float red[256];

    int row = blockIdx.x;            // (b*H + h)*T + i
    int i  = row % Tn;
    int bh = row / Tn;
    int h  = bh % Hn;
    int b  = bh / Hn;
    int tid = threadIdx.x;

    float qws[Sn];
    const float* qp = qw + ((long long)bh * Tn + i) * Sn;
    #pragma unroll
    for (int s = 0; s < Sn; ++s) qws[s] = qp[s];

    const float invT = scal[1];
    const float* kb = kwa + (long long)bh * Sn * Tn;

    float l[JPT];
    float lmax = -3.402823466e+38f;
    #pragma unroll
    for (int u = 0; u < JPT; ++u) {
        int j = tid + u * 256;
        float acc = 0.0f;
        #pragma unroll
        for (int s = 0; s < Sn; ++s) acc += qws[s] * kb[(long long)s * Tn + j];
        acc = acc * invT + 1e-8f;
        l[u] = acc;
        lmax = fmaxf(lmax, acc);
    }
    red[tid] = lmax;
    __syncthreads();
    for (int o = 128; o > 0; o >>= 1) {
        if (tid < o) red[tid] = fmaxf(red[tid], red[tid + o]);
        __syncthreads();
    }
    float rmax = red[0];
    __syncthreads();

    float lsum = 0.0f;
    #pragma unroll
    for (int u = 0; u < JPT; ++u) {
        l[u] = __expf(l[u] - rmax);
        lsum += l[u];
    }
    red[tid] = lsum;
    __syncthreads();
    for (int o = 128; o > 0; o >>= 1) {
        if (tid < o) red[tid] += red[tid + o];
        __syncthreads();
    }
    float inv = 1.0f / red[0];

    float* ao = attn_out + ((long long)(b * Tn + i) * Tn) * Hn + h;
    unsigned short* ab = attn_bf + ((long long)bh * Tn + i) * Tn;
    #pragma unroll
    for (int u = 0; u < JPT; ++u) {
        int j = tid + u * 256;
        float p = l[u] * inv;
        ao[(long long)j * Hn] = p;
        ab[j] = f2bf_bits(p);
    }
}

// ---------------------------------------------------------------------------

extern "C" void kernel_launch(void* const* d_in, const int* in_sizes, int n_in,
                              void* d_out, int out_size, void* d_ws, size_t ws_size,
                              hipStream_t stream) {
    (void)in_sizes; (void)n_in; (void)out_size; (void)ws_size;

    const float* x        = (const float*)d_in[0];
    const float* W_qkv    = (const float*)d_in[1];
    const float* W_out    = (const float*)d_in[2];
    const float* base_c   = (const float*)d_in[3];
    const float* deltas   = (const float*)d_in[4];
    const float* log_sc   = (const float*)d_in[5];
    const float* log_am   = (const float*)d_in[6];
    const float* movement = (const float*)d_in[7];
    const float* temp     = (const float*)d_in[8];

    float* out_proj = (float*)d_out;                                   // [B,T,DIM]
    float* attn_out = (float*)d_out + (long long)MTn * DIMn;           // [B,T,T,H]

    // ---- workspace carve-up (aligned to 256B) ----
    size_t off = 0;
    auto alloc = [&](size_t bytes) -> char* {
        char* p = (char*)d_ws + off;
        off += (bytes + 255) & ~(size_t)255;
        return p;
    };
    float*          scal     = (float*)alloc(16 * sizeof(float));
    float*          centers  = (float*)alloc((size_t)Hn * Sn * HDn * 4);
    float*          c_sq     = (float*)alloc((size_t)Hn * Sn * 4);
    float*          inv_var  = (float*)alloc((size_t)Hn * Sn * 4);
    float*          amps     = (float*)alloc((size_t)Hn * Sn * 4);
    unsigned short* x_bf     = (unsigned short*)alloc((size_t)MTn * DIMn * 2);
    unsigned short* wqkv_bf  = (unsigned short*)alloc((size_t)N3n * DIMn * 2);
    unsigned short* wout_bf  = (unsigned short*)alloc((size_t)DIMn * DIMn * 2);
    float*          qkv_f32  = (float*)alloc((size_t)MTn * N3n * 4);
    float*          qw       = (float*)alloc((size_t)Bn * Hn * Tn * Sn * 4);
    float*          kwa      = (float*)alloc((size_t)Bn * Hn * Sn * Tn * 4);
    unsigned short* vt_bf    = (unsigned short*)alloc((size_t)Bn * Hn * HDn * Tn * 2);
    unsigned short* attn_bf  = (unsigned short*)alloc((size_t)Bn * Hn * Tn * Tn * 2);
    float*          ao_f32   = (float*)alloc((size_t)MTn * DIMn * 4);
    unsigned short* ao_bf    = (unsigned short*)alloc((size_t)MTn * DIMn * 2);

    // 1) prep (centers / inv_var / c_sq / amps / 1/temp)
    prep_kernel<<<1, 256, 0, stream>>>(base_c, deltas, log_sc, log_am,
                                       movement, temp,
                                       centers, c_sq, inv_var, amps, scal);

    // 2) fp32 -> bf16 for GEMM operands
    {
        long long n = (long long)MTn * DIMn;
        f32_to_bf16_kernel<<<(int)((n + 255) / 256), 256, 0, stream>>>(x, x_bf, n);
        n = (long long)N3n * DIMn;
        f32_to_bf16_kernel<<<(int)((n + 255) / 256), 256, 0, stream>>>(W_qkv, wqkv_bf, n);
        n = (long long)DIMn * DIMn;
        f32_to_bf16_kernel<<<(int)((n + 255) / 256), 256, 0, stream>>>(W_out, wout_bf, n);
    }

    // 3) qkv = x @ W_qkv^T            (M=4096, N=3072, K=1024)
    gemm_nt_bf16<<<dim3(N3n / 64, MTn / 128, 1), 128, 0, stream>>>(
        (const __bf16*)x_bf, DIMn, 0,
        (const __bf16*)wqkv_bf, DIMn, 0,
        qkv_f32, N3n, 1, 0, 0,
        MTn, N3n, DIMn);

    // 4) splat weights (q -> qw, k*amps -> kwa)
    splat_kernel<<<(Bn * Tn * Hn) / 4, 128, 0, stream>>>(
        qkv_f32, centers, c_sq, inv_var, amps, qw, kwa);

    // 5) v transpose + bf16 (LDS tiled)
    vt_transpose_kernel<<<dim3(Tn / 32, (HDn / 32) * Bn * Hn, 1),
                          dim3(32, 8, 1), 0, stream>>>(qkv_f32, vt_bf);

    // 6) fused rank-S logits + softmax  -> attn (fp32 to d_out, bf16 to ws)
    attn_softmax_kernel<<<Bn * Hn * Tn, 256, 0, stream>>>(
        qw, kwa, scal, attn_out, attn_bf);

    // 7) per-(b,h): ao = attn @ v      (16 batches, M=2048, N=128, K=2048)
    gemm_nt_bf16<<<dim3(HDn / 64, Tn / 128, Bn * Hn), 128, 0, stream>>>(
        (const __bf16*)attn_bf, Tn, (long long)Tn * Tn,
        (const __bf16*)vt_bf,   Tn, (long long)HDn * Tn,
        ao_f32, DIMn, Hn, (long long)Tn * DIMn, (long long)HDn,
        Tn, HDn, Tn);

    // 8) ao -> bf16
    {
        long long n = (long long)MTn * DIMn;
        f32_to_bf16_kernel<<<(int)((n + 255) / 256), 256, 0, stream>>>(ao_f32, ao_bf, n);
    }

    // 9) out = ao @ W_out^T  -> d_out  (M=4096, N=1024, K=1024)
    gemm_nt_bf16<<<dim3(DIMn / 64, MTn / 128, 1), 128, 0, stream>>>(
        (const __bf16*)ao_bf, DIMn, 0,
        (const __bf16*)wout_bf, DIMn, 0,
        out_proj, DIMn, 1, 0, 0,
        MTn, DIMn, DIMn);
}
